// TransformerBlock_60911226191905
// MI455X (gfx1250) — matrix operand order repaired and checked by reading, not compiled, not run on hardware
//
#include <hip/hip_runtime.h>
#include <stdint.h>

#define EMBED   384
#define HEADS   6
#define HDIM    64
#define MLP_HID 1536
#define BATCH   16
#define SEQ     1024
#define MTOT    (BATCH * SEQ)   // 16384 rows

typedef __attribute__((ext_vector_type(16))) __bf16 v16bf;
typedef __attribute__((ext_vector_type(8)))  float  v8f;
typedef __attribute__((ext_vector_type(4)))  int    v4i;

union AFrag { v16bf v; uint4 q[2]; };

#ifndef __has_builtin
#define __has_builtin(x) 0
#endif
#if __has_builtin(__builtin_amdgcn_global_load_async_to_lds_b128) && \
    __has_builtin(__builtin_amdgcn_s_wait_asynccnt)
#define HAVE_ASYNC 1
#else
#define HAVE_ASYNC 0
#endif

typedef __attribute__((address_space(1))) v4i gv4i;   // global v4i
typedef __attribute__((address_space(3))) v4i lv4i;   // LDS v4i

// 16-byte global -> LDS copy; async (ASYNCcnt) when the builtin exists.
__device__ __forceinline__ void cp16(const unsigned short* g, unsigned short* l) {
#if HAVE_ASYNC
  __builtin_amdgcn_global_load_async_to_lds_b128(
      (gv4i*)(unsigned long long)g,
      (lv4i*)(unsigned)(unsigned long long)l, 0, 0);
#else
  *(uint4*)l = *(const uint4*)g;
#endif
}
__device__ __forceinline__ void cp_wait() {
#if HAVE_ASYNC
  __builtin_amdgcn_s_wait_asynccnt(0);
#endif
}

__device__ __forceinline__ unsigned short f2bf(float f) {
  union { float f; unsigned u; } c; c.f = f;
  unsigned u = c.u;
  return (unsigned short)((u + 0x7FFFu + ((u >> 16) & 1u)) >> 16);  // RNE
}

// ---------------------------------------------------------------------------
// fp32 (KxN) -> bf16 transposed (NxK) weight conversion
// ---------------------------------------------------------------------------
__global__ void convert_transpose_kernel(const float* __restrict__ in,
                                         unsigned short* __restrict__ out,
                                         int K, int N) {
  int i = blockIdx.x * blockDim.x + threadIdx.x;
  if (i < K * N) {
    int k = i / N, n = i % N;
    out[(size_t)n * K + k] = f2bf(in[i]);
  }
}

// ---------------------------------------------------------------------------
// LayerNorm over C=384, one wave32 per row, 12 elems/lane
// ---------------------------------------------------------------------------
__global__ __launch_bounds__(256) void layernorm_bf16_kernel(
    const float* __restrict__ x, const float* __restrict__ g,
    const float* __restrict__ b, unsigned short* __restrict__ out) {
  const int lane = threadIdx.x & 31;
  const int wave = threadIdx.x >> 5;
  const int row  = blockIdx.x * 8 + wave;
  const float* xr = x + (size_t)row * EMBED;

  float v[12];
  float s = 0.f, sq = 0.f;
#pragma unroll
  for (int i = 0; i < 12; ++i) {
    float t = xr[lane + i * 32];
    v[i] = t; s += t; sq += t * t;
  }
#pragma unroll
  for (int m = 16; m >= 1; m >>= 1) {
    s  += __shfl_xor(s,  m, 32);
    sq += __shfl_xor(sq, m, 32);
  }
  const float mu  = s * (1.f / EMBED);
  const float var = sq * (1.f / EMBED) - mu * mu;
  const float rs  = rsqrtf(var + 1e-5f);

  unsigned short* orow = out + (size_t)row * EMBED;
#pragma unroll
  for (int i = 0; i < 12; ++i) {
    int c = lane + i * 32;
    orow[c] = f2bf((v[i] - mu) * rs * g[c] + b[c]);
  }
}

// ---------------------------------------------------------------------------
// Tiled WMMA GEMM: D = A(MxK,bf16) @ W(KxN,bf16; stored transposed NxK) + bias
// block tile 128x64, BK=32, 8 waves in 4x2, each wave 32x32 (2x2 wmma tiles)
// Double-buffered LDS; staging via async global->LDS B128 when available.
// ---------------------------------------------------------------------------
#define BM 128
#define BN 64
#define BK 32
#define LPAD 8
#define LSTR (BK + LPAD)   // 40 elems = 80B row stride (16B aligned)

enum { EPI_QKV = 0, EPI_GELU = 1, EPI_RES = 2 };

template <int EPI>
__global__ __launch_bounds__(256) void gemm_bf16_kernel(
    const unsigned short* __restrict__ A, const unsigned short* __restrict__ Wt,
    const float* __restrict__ bias, const float* __restrict__ res,
    void* __restrict__ outp, int M, int N, int K) {
  __shared__ unsigned short As[2][BM * LSTR];
  __shared__ unsigned short Bs[2][BN * LSTR];   // [n][k] rows (Wt is NxK)

  const int tid  = threadIdx.x;
  const int lane = tid & 31;
  const int wave = tid >> 5;
  const int wm   = wave & 3;    // row group (32 rows each)
  const int wn   = wave >> 2;   // col group (32 cols each)
  const int bm0  = blockIdx.y * BM;
  const int bn0  = blockIdx.x * BN;

  const v8f vzero = {0.f, 0.f, 0.f, 0.f, 0.f, 0.f, 0.f, 0.f};
  v8f acc[2][2];
  acc[0][0] = vzero; acc[0][1] = vzero; acc[1][0] = vzero; acc[1][1] = vzero;

  const int arow  = tid >> 1;        // 0..127, A: 2 x 16 elems per row
  const int ahalf = tid & 1;
  const int brow  = tid >> 2;        // 0..63,  B: 4 x 8 elems per row
  const int bseg  = tid & 3;

  auto stage = [&](int k0, int buf) {
    const unsigned short* ga = A + (size_t)(bm0 + arow) * K + k0 + ahalf * 16;
    unsigned short* la = &As[buf][arow * LSTR + ahalf * 16];
    cp16(ga, la);
    cp16(ga + 8, la + 8);
    const unsigned short* gb = Wt + (size_t)(bn0 + brow) * K + k0 + bseg * 8;
    cp16(gb, &Bs[buf][brow * LSTR + bseg * 8]);
  };

  stage(0, 0);
  cp_wait();
  __syncthreads();

  int cur = 0;
  for (int k0 = 0; k0 < K; k0 += BK) {
    const int nxt = cur ^ 1;
    if (k0 + BK < K) stage(k0 + BK, nxt);   // prefetch next tile into other buf

    AFrag af[2], bf[2];
#pragma unroll
    for (int mi = 0; mi < 2; ++mi) {
      const unsigned short* src =
          &As[cur][(wm * 32 + mi * 16 + (lane & 15)) * LSTR + ((lane >> 4) * 8)];
      af[mi].q[0] = *(const uint4*)(src);
      af[mi].q[1] = *(const uint4*)(src + 16);
    }
#pragma unroll
    for (int ni = 0; ni < 2; ++ni) {
      const unsigned short* src =
          &Bs[cur][(wn * 32 + ni * 16 + (lane & 15)) * LSTR + ((lane >> 4) * 8)];
      bf[ni].q[0] = *(const uint4*)(src);
      bf[ni].q[1] = *(const uint4*)(src + 16);
    }
#pragma unroll
    for (int mi = 0; mi < 2; ++mi)
#pragma unroll
      for (int ni = 0; ni < 2; ++ni)
        acc[mi][ni] = __builtin_amdgcn_wmma_f32_16x16x32_bf16(
            false, af[mi].v, false, bf[ni].v, (short)0, acc[mi][ni], false, false);

    cp_wait();        // own async writes for nxt landed
    __syncthreads();  // everyone done reading cur + staging nxt
    cur = nxt;
  }

  // --- epilogue ---
#pragma unroll
  for (int mi = 0; mi < 2; ++mi) {
#pragma unroll
    for (int ni = 0; ni < 2; ++ni) {
      const int gn = bn0 + wn * 32 + ni * 16 + (lane & 15);
      const float bv = bias[gn];
#pragma unroll
      for (int r = 0; r < 8; ++r) {
        const int gm = bm0 + wm * 32 + mi * 16 + ((lane >> 4) * 8) + r;
        float val = acc[mi][ni][r] + bv;
        if constexpr (EPI == EPI_QKV) {
          const int t  = gn / EMBED;      // 0=q 1=k 2=v
          const int hh = (gn % EMBED) / HDIM;
          const int d  = gn % HDIM;
          const int bb = gm / SEQ, nn = gm % SEQ;
          unsigned short* base = (unsigned short*)outp;
          size_t idx;
          if (t == 2)   // V stored transposed: [b][h][d][n]
            idx = 2 * (size_t)MTOT * EMBED +
                  (((size_t)(bb * HEADS + hh) * HDIM + d) * SEQ + nn);
          else          // Q,K: [b][h][n][d]
            idx = (size_t)t * ((size_t)MTOT * EMBED) +
                  (((size_t)(bb * HEADS + hh) * SEQ + nn) * HDIM + d);
          base[idx] = f2bf(val);
        } else if constexpr (EPI == EPI_GELU) {
          const float gl = 0.5f * val * (1.0f + erff(val * 0.70710678f));
          ((unsigned short*)outp)[(size_t)gm * N + gn] = f2bf(gl);
        } else {  // EPI_RES: bias + residual -> f32
          ((float*)outp)[(size_t)gm * N + gn] = val + res[(size_t)gm * N + gn];
        }
      }
    }
  }
}

// ---------------------------------------------------------------------------
// Flash attention: block = (b,h, 128-query strip); wave owns 16 query rows,
// streams 32 keys/iter. Q,K bf16 [B,H,N,D]; V bf16 [B,H,D,N] (pre-transposed).
// Row-sums of P computed with an extra WMMA against a ones matrix.
// ---------------------------------------------------------------------------
__global__ __launch_bounds__(256) void attention_kernel(
    const unsigned short* __restrict__ Q, const unsigned short* __restrict__ K,
    const unsigned short* __restrict__ V, unsigned short* __restrict__ ctx) {
  __shared__ unsigned short Plds[8 * 16 * LSTR];  // per-wave 16x32 P tile
  __shared__ unsigned short Vt[64 * LSTR];        // V^T chunk: [d][key]

  const int lane = threadIdx.x & 31;
  const int wave = threadIdx.x >> 5;
  const int bh   = blockIdx.x >> 3;   // 0..95
  const int qt   = blockIdx.x & 7;
  const int q0   = qt * 128 + wave * 16;

  const unsigned short* Qb = Q + (size_t)bh * SEQ * HDIM;
  const unsigned short* Kb = K + (size_t)bh * SEQ * HDIM;
  const unsigned short* Vb = V + (size_t)bh * HDIM * SEQ;  // [d][n]

  AFrag qf[2];
  {
    const unsigned short* qrow = Qb + (size_t)(q0 + (lane & 15)) * HDIM;
    const int kb = (lane >> 4) * 8;
    qf[0].q[0] = *(const uint4*)(qrow + kb);
    qf[0].q[1] = *(const uint4*)(qrow + kb + 16);
    qf[1].q[0] = *(const uint4*)(qrow + 32 + kb);
    qf[1].q[1] = *(const uint4*)(qrow + 32 + kb + 16);
  }
  AFrag onesf;   // B matrix of all 1.0 (bf16 0x3F80) for row-sum WMMA
  onesf.q[0] = make_uint4(0x3F803F80u, 0x3F803F80u, 0x3F803F80u, 0x3F803F80u);
  onesf.q[1] = onesf.q[0];

  const v8f vzero = {0.f, 0.f, 0.f, 0.f, 0.f, 0.f, 0.f, 0.f};
  v8f o[4]; o[0] = vzero; o[1] = vzero; o[2] = vzero; o[3] = vzero;
  float mrow[8], lrow[8];
#pragma unroll
  for (int r = 0; r < 8; ++r) { mrow[r] = -3.0e38f; lrow[r] = 0.f; }

  unsigned short* myP = Plds + wave * 16 * LSTR;
  const float scale = 0.125f;  // 1/sqrt(64)

  const int vd   = threadIdx.x >> 2;      // 0..63  (V^T row = feature d)
  const int vseg = threadIdx.x & 3;       // 16B segment along keys

  for (int j0 = 0; j0 < SEQ; j0 += 32) {
    // stage V^T chunk: contiguous 16B copies (async when available)
    cp16(Vb + (size_t)vd * SEQ + j0 + vseg * 8, &Vt[vd * LSTR + vseg * 8]);
    cp_wait();
    __syncthreads();

    if (j0 + 32 < SEQ) {  // prefetch next K / V chunks into cache
      __builtin_prefetch(Kb + (size_t)(j0 + 32 + lane) * HDIM, 0, 0);
      __builtin_prefetch(Vb + (size_t)(lane * 2) * SEQ + j0 + 32, 0, 0);
    }

    // S = Q @ K^T for 32 keys -> two 16x16 acc tiles
    v8f s0 = vzero, s1 = vzero;
#pragma unroll
    for (int c = 0; c < 2; ++c) {
      AFrag kf0, kf1;
      const int kb = c * 32 + (lane >> 4) * 8;
      const unsigned short* kr0 = Kb + (size_t)(j0 + (lane & 15)) * HDIM + kb;
      const unsigned short* kr1 = Kb + (size_t)(j0 + 16 + (lane & 15)) * HDIM + kb;
      kf0.q[0] = *(const uint4*)(kr0); kf0.q[1] = *(const uint4*)(kr0 + 16);
      kf1.q[0] = *(const uint4*)(kr1); kf1.q[1] = *(const uint4*)(kr1 + 16);
      s0 = __builtin_amdgcn_wmma_f32_16x16x32_bf16(false, qf[c].v, false, kf0.v,
                                                   (short)0, s0, false, false);
      s1 = __builtin_amdgcn_wmma_f32_16x16x32_bf16(false, qf[c].v, false, kf1.v,
                                                   (short)0, s1, false, false);
    }

    // online softmax: max via half-wave shuffles; sums via WMMA below
    float alpha[8];
#pragma unroll
    for (int r = 0; r < 8; ++r) {
      float a = s0[r] * scale, b = s1[r] * scale;
      float mx = fmaxf(a, b);
#pragma unroll
      for (int m = 8; m >= 1; m >>= 1) mx = fmaxf(mx, __shfl_xor(mx, m, 32));
      const float newm = fmaxf(mrow[r], mx);
      const float p0 = __expf(a - newm), p1 = __expf(b - newm);
      alpha[r] = __expf(mrow[r] - newm);
      lrow[r] *= alpha[r];
      mrow[r]  = newm;
      const int row = ((lane >> 4) * 8) + r;
      myP[row * LSTR + (lane & 15)]      = f2bf(p0);
      myP[row * LSTR + 16 + (lane & 15)] = f2bf(p1);
    }
#pragma unroll
    for (int dt = 0; dt < 4; ++dt)
#pragma unroll
      for (int r = 0; r < 8; ++r) o[dt][r] *= alpha[r];

    // O += P @ V ; row-sum(P) = P @ ones (broadcast across columns)
    AFrag pf;
    {
      const unsigned short* src = myP + (lane & 15) * LSTR + ((lane >> 4) * 8);
      pf.q[0] = *(const uint4*)(src);
      pf.q[1] = *(const uint4*)(src + 16);
    }
    v8f sacc = __builtin_amdgcn_wmma_f32_16x16x32_bf16(false, pf.v, false, onesf.v,
                                                       (short)0, vzero, false, false);
#pragma unroll
    for (int dt = 0; dt < 4; ++dt) {
      AFrag vf;
      const unsigned short* src = &Vt[(dt * 16 + (lane & 15)) * LSTR + ((lane >> 4) * 8)];
      vf.q[0] = *(const uint4*)(src);
      vf.q[1] = *(const uint4*)(src + 16);
      o[dt] = __builtin_amdgcn_wmma_f32_16x16x32_bf16(false, pf.v, false, vf.v,
                                                      (short)0, o[dt], false, false);
    }
#pragma unroll
    for (int r = 0; r < 8; ++r) lrow[r] += sacc[r];

    __syncthreads();  // all waves done with Vt before restage
  }

  // normalize and write ctx as [B, N, H*D] bf16
  const int bb = bh / HEADS, hh = bh % HEADS;
#pragma unroll
  for (int dt = 0; dt < 4; ++dt) {
#pragma unroll
    for (int r = 0; r < 8; ++r) {
      const int qrow = q0 + ((lane >> 4) * 8) + r;
      const int d = dt * 16 + (lane & 15);
      const float val = o[dt][r] / lrow[r];
      ctx[((size_t)bb * SEQ + qrow) * EMBED + hh * HDIM + d] = f2bf(val);
    }
  }
}

// ---------------------------------------------------------------------------
// launch
// ---------------------------------------------------------------------------
extern "C" void kernel_launch(void* const* d_in, const int* in_sizes, int n_in,
                              void* d_out, int out_size, void* d_ws, size_t ws_size,
                              hipStream_t stream) {
  (void)in_sizes; (void)n_in; (void)out_size; (void)ws_size;
  const float* x      = (const float*)d_in[0];
  const float* n1g    = (const float*)d_in[1];
  const float* n1b    = (const float*)d_in[2];
  const float* qkv_w  = (const float*)d_in[3];
  const float* qkv_b  = (const float*)d_in[4];
  const float* proj_w = (const float*)d_in[5];
  const float* proj_b = (const float*)d_in[6];
  const float* n2g    = (const float*)d_in[7];
  const float* n2b    = (const float*)d_in[8];
  const float* fc1_w  = (const float*)d_in[9];
  const float* fc1_b  = (const float*)d_in[10];
  const float* fc2_w  = (const float*)d_in[11];
  const float* fc2_b  = (const float*)d_in[12];

  const size_t S_ACT = (size_t)MTOT * EMBED;
  unsigned short* hA    = (unsigned short*)d_ws;            // LN output (reused)
  unsigned short* wqkv  = hA    + S_ACT;                    // transposed bf16 weights
  unsigned short* wproj = wqkv  + (size_t)EMBED * 3 * EMBED;
  unsigned short* wfc1  = wproj + (size_t)EMBED * EMBED;
  unsigned short* wfc2  = wfc1  + (size_t)EMBED * MLP_HID;
  unsigned short* qbuf  = wfc2  + (size_t)MLP_HID * EMBED;  // q,k,v contiguous
  unsigned short* ctx   = qbuf + 3 * S_ACT;
  unsigned short* hmlp  = qbuf;                             // reuse (exact fit)
  float* xres = (float*)(ctx + S_ACT);

  // 1) weight conversion fp32 -> bf16, transposed to NxK
  {
    const int nq = EMBED * 3 * EMBED, np = EMBED * EMBED, n1 = EMBED * MLP_HID;
    convert_transpose_kernel<<<(nq + 255) / 256, 256, 0, stream>>>(qkv_w, wqkv, EMBED, 3 * EMBED);
    convert_transpose_kernel<<<(np + 255) / 256, 256, 0, stream>>>(proj_w, wproj, EMBED, EMBED);
    convert_transpose_kernel<<<(n1 + 255) / 256, 256, 0, stream>>>(fc1_w, wfc1, EMBED, MLP_HID);
    convert_transpose_kernel<<<(n1 + 255) / 256, 256, 0, stream>>>(fc2_w, wfc2, MLP_HID, EMBED);
  }

  // 2) LN1
  layernorm_bf16_kernel<<<MTOT / 8, 256, 0, stream>>>(x, n1g, n1b, hA);

  // 3) QKV GEMM -> bf16 Q,K [B,H,N,D], V [B,H,D,N]
  gemm_bf16_kernel<EPI_QKV><<<dim3((3 * EMBED) / BN, MTOT / BM), 256, 0, stream>>>(
      hA, wqkv, qkv_b, nullptr, qbuf, MTOT, 3 * EMBED, EMBED);

  // 4) attention -> ctx bf16 [B,N,C]
  attention_kernel<<<BATCH * HEADS * (SEQ / 128), 256, 0, stream>>>(
      qbuf, qbuf + S_ACT, qbuf + 2 * S_ACT, ctx);

  // 5) proj GEMM + bias + residual(x) -> xres f32
  gemm_bf16_kernel<EPI_RES><<<dim3(EMBED / BN, MTOT / BM), 256, 0, stream>>>(
      ctx, wproj, proj_b, x, xres, MTOT, EMBED, EMBED);

  // 6) LN2
  layernorm_bf16_kernel<<<MTOT / 8, 256, 0, stream>>>(xres, n2g, n2b, hA);

  // 7) FC1 GEMM + bias + exact GELU -> hmlp bf16
  gemm_bf16_kernel<EPI_GELU><<<dim3(MLP_HID / BN, MTOT / BM), 256, 0, stream>>>(
      hA, wfc1, fc1_b, nullptr, hmlp, MTOT, MLP_HID, EMBED);

  // 8) FC2 GEMM + bias + residual(xres) -> d_out f32
  gemm_bf16_kernel<EPI_RES><<<dim3(EMBED / BN, MTOT / BM), 256, 0, stream>>>(
      hmlp, wfc2, fc2_b, xres, (float*)d_out, MTOT, EMBED, MLP_HID);
}
